// MPSUMambaDecoder_78675210928767
// MI455X (gfx1250) — compile-verified, matrix operand order checked by reading, not run on hardware
//
#include <hip/hip_runtime.h>

// Problem dimensions (fixed by the reference).
#define B_   8
#define L_   512
#define D_   256
#define DI_  512
#define N_   8
#define R_   16
#define K_   4
#define DFF_ 1024
#define V_   9

typedef __attribute__((ext_vector_type(16))) __bf16 v16bf;
typedef __attribute__((ext_vector_type(4)))  __bf16 bf16x4;
typedef __attribute__((ext_vector_type(8)))  float  v8f;
typedef int v4i __attribute__((__vector_size__(4 * sizeof(int))));

#ifndef __has_builtin
#define __has_builtin(x) 0
#endif
#if __has_builtin(__builtin_amdgcn_global_load_async_to_lds_b128) && \
    __has_builtin(__builtin_amdgcn_s_wait_asynccnt)
#define HAVE_ASYNC_LDS 1
// Builtin expects (v4i addrspace(1)* src, v4i addrspace(3)* dst, imm off, imm cpol).
#define AS1P(p) ((__attribute__((address_space(1))) v4i*)(unsigned long long)(p))
#define AS3P(p) ((__attribute__((address_space(3))) v4i*)(unsigned)(unsigned long long)(p))
#else
#define HAVE_ASYNC_LDS 0
#endif

// f32 -> bf16 round-to-nearest-even via bit manipulation.
__device__ inline __bf16 f2bf(float f) {
  union { float f; unsigned u; } a; a.f = f;
  unsigned r = (a.u + 0x7FFFu + ((a.u >> 16) & 1u)) >> 16;
  union { unsigned short s; __bf16 b; } o; o.s = (unsigned short)r;
  return o.b;
}

__device__ inline float siluf(float x) { return x / (1.f + __expf(-x)); }

__device__ inline float apply_act(float v, int act) {
  if (act == 1) return v > 0.f ? v : 0.f;                 // relu
  if (act == 2) return v > 20.f ? v : log1pf(__expf(v));  // softplus
  return v;
}

#define BM 64
#define BN 128
#define BK 64
#define PAD 8
#define LDK (BK + PAD)

// ---------------------------------------------------------------------------
// Tile staging for the fast (fully in-bounds) path: f32 -> bf16, vectorized.
// ---------------------------------------------------------------------------
template <int TRANSB>
__device__ __forceinline__ void mpsu_stage_tiles(
    const float* __restrict__ A, const float* __restrict__ Bg,
    int lda, int ldb, int m0, int n0, int k0, int tid,
    __bf16 (*As)[LDK], __bf16 (*Bs)[LDK])
{
#pragma unroll
  for (int i = 0; i < (BM * BK / 4) / 256; ++i) {   // 4 iterations
    int t  = tid + i * 256;
    int m  = t >> 4;             // BK/4 = 16
    int k4 = (t & 15) << 2;
    float4 v = *reinterpret_cast<const float4*>(
        &A[(long long)(m0 + m) * lda + k0 + k4]);
    bf16x4 o = { f2bf(v.x), f2bf(v.y), f2bf(v.z), f2bf(v.w) };
    *reinterpret_cast<bf16x4*>(&As[m][k4]) = o;
  }
  if (TRANSB) {
#pragma unroll
    for (int i = 0; i < (BN * BK / 4) / 256; ++i) { // 8 iterations
      int t  = tid + i * 256;
      int n  = t >> 4;
      int k4 = (t & 15) << 2;
      float4 v = *reinterpret_cast<const float4*>(
          &Bg[(long long)(n0 + n) * ldb + k0 + k4]);
      bf16x4 o = { f2bf(v.x), f2bf(v.y), f2bf(v.z), f2bf(v.w) };
      *reinterpret_cast<bf16x4*>(&Bs[n][k4]) = o;
    }
  } else {
#pragma unroll
    for (int i = 0; i < (BK * BN / 4) / 256; ++i) { // 8 iterations
      int t  = tid + i * 256;
      int kk = t >> 5;             // BN/4 = 32
      int n4 = (t & 31) << 2;
      float4 v = *reinterpret_cast<const float4*>(
          &Bg[(long long)(k0 + kk) * ldb + n0 + n4]);
      Bs[n4 + 0][kk] = f2bf(v.x);
      Bs[n4 + 1][kk] = f2bf(v.y);
      Bs[n4 + 2][kk] = f2bf(v.z);
      Bs[n4 + 3][kk] = f2bf(v.w);
    }
  }
}

// ---------------------------------------------------------------------------
// FAST batched GEMM (M%64==0, N%128==0, K%64==0 — every big GEMM here).
// C[M,N] = act(alpha * A @ B(^T) + bias). Double-buffered LDS: stage tile t+1
// while 8 WMMAs consume tile t; one barrier per K stage; unguarded stores.
// ---------------------------------------------------------------------------
template <int TRANSB>
__global__ __launch_bounds__(256)
void mpsu_gemm_fast(const float* __restrict__ A, const float* __restrict__ Bg,
                    float* __restrict__ C, const float* __restrict__ bias,
                    int M, int N, int K, int lda, int ldb, int ldc,
                    long long sA, long long sB, long long sC,
                    float alpha, int act)
{
  __shared__ __bf16 As[2][BM][LDK];
  __shared__ __bf16 Bs[2][BN][LDK];

  const int bz = blockIdx.z;
  A += bz * sA; Bg += bz * sB; C += bz * sC;

  const int m0   = blockIdx.y * BM;
  const int n0   = blockIdx.x * BN;
  const int tid  = threadIdx.x;
  const int lane = tid & 31;
  const int wave = tid >> 5;
  const int wm   = wave >> 2;  // 2 waves along M
  const int wn   = wave & 3;   // 4 waves along N
  const int row  = lane & 15;
  const int half = lane >> 4;

  v8f acc00 = {}, acc01 = {}, acc10 = {}, acc11 = {};

  mpsu_stage_tiles<TRANSB>(A, Bg, lda, ldb, m0, n0, 0, tid, As[0], Bs[0]);
  __syncthreads();

  const int nk = K / BK;
  for (int t = 0; t < nk; ++t) {
    const int cur = t & 1;
    if (t + 1 < nk)
      mpsu_stage_tiles<TRANSB>(A, Bg, lda, ldb, m0, n0, (t + 1) * BK, tid,
                               As[cur ^ 1], Bs[cur ^ 1]);
#pragma unroll
    for (int s = 0; s < 2; ++s) {
      const int kb = s * 32 + half * 16;
      v16bf a0, a1, b0, b1;
      const __bf16* pa0 = &As[cur][wm * 32 + row][kb];
      const __bf16* pa1 = &As[cur][wm * 32 + 16 + row][kb];
      const __bf16* pb0 = &Bs[cur][wn * 32 + row][kb];
      const __bf16* pb1 = &Bs[cur][wn * 32 + 16 + row][kb];
#pragma unroll
      for (int i = 0; i < 16; ++i) {
        a0[i] = pa0[i]; a1[i] = pa1[i]; b0[i] = pb0[i]; b1[i] = pb1[i];
      }
      acc00 = __builtin_amdgcn_wmma_f32_16x16x32_bf16(false, a0, false, b0,
                                                      (short)0, acc00, false, false);
      acc01 = __builtin_amdgcn_wmma_f32_16x16x32_bf16(false, a0, false, b1,
                                                      (short)0, acc01, false, false);
      acc10 = __builtin_amdgcn_wmma_f32_16x16x32_bf16(false, a1, false, b0,
                                                      (short)0, acc10, false, false);
      acc11 = __builtin_amdgcn_wmma_f32_16x16x32_bf16(false, a1, false, b1,
                                                      (short)0, acc11, false, false);
    }
    __syncthreads();
  }

  const int gmB = m0 + wm * 32 + half * 8;
  const int cB  = n0 + wn * 32 + row;
  const float b0v = bias ? bias[cB] : 0.f;
  const float b1v = bias ? bias[cB + 16] : 0.f;
#pragma unroll
  for (int r = 0; r < 8; ++r) {
    const long long r0 = (long long)(gmB + r) * ldc;
    const long long r1 = (long long)(gmB + 16 + r) * ldc;
    C[r0 + cB]      = apply_act(acc00[r] * alpha + b0v, act);
    C[r0 + cB + 16] = apply_act(acc01[r] * alpha + b1v, act);
    C[r1 + cB]      = apply_act(acc10[r] * alpha + b0v, act);
    C[r1 + cB + 16] = apply_act(acc11[r] * alpha + b1v, act);
  }
}

// ---------------------------------------------------------------------------
// GENERAL batched GEMM (bounds-checked; used only for the two small odd-shape
// GEMMs: W_xp (N=32) and W_dt (K=16)).
// ---------------------------------------------------------------------------
__global__ __launch_bounds__(256)
void mpsu_gemm_wmma(const float* __restrict__ A, const float* __restrict__ Bg,
                    float* __restrict__ C, const float* __restrict__ bias,
                    int M, int N, int K, int lda, int ldb, int ldc,
                    long long sA, long long sB, long long sC,
                    int transB, float alpha, int act)
{
  __shared__ __bf16 As[BM][LDK];
  __shared__ __bf16 Bs[BN][LDK];

  const int bz = blockIdx.z;
  A += bz * sA; Bg += bz * sB; C += bz * sC;

  const int m0   = blockIdx.y * BM;
  const int n0   = blockIdx.x * BN;
  const int tid  = threadIdx.x;
  const int lane = tid & 31;
  const int wave = tid >> 5;
  const int wm   = wave >> 2;
  const int wn   = wave & 3;
  const int row  = lane & 15;
  const int half = lane >> 4;

  v8f acc00 = {}, acc01 = {}, acc10 = {}, acc11 = {};

  for (int k0 = 0; k0 < K; k0 += BK) {
#pragma unroll
    for (int i = 0; i < (BM * BK) / 256; ++i) {
      int t = tid + i * 256;
      int m = t >> 6, kk = t & 63;
      int gm = m0 + m, gk = k0 + kk;
      float v = 0.f;
      if (gm < M && gk < K) v = A[(long long)gm * lda + gk];
      As[m][kk] = f2bf(v);
    }
#pragma unroll
    for (int i = 0; i < (BN * BK) / 256; ++i) {
      int t = tid + i * 256;
      int n = t >> 6, kk = t & 63;
      int gn = n0 + n, gk = k0 + kk;
      float v = 0.f;
      if (gn < N && gk < K)
        v = transB ? Bg[(long long)gn * ldb + gk]
                   : Bg[(long long)gk * ldb + gn];
      Bs[n][kk] = f2bf(v);
    }
    __syncthreads();

#pragma unroll
    for (int s = 0; s < 2; ++s) {
      const int kb = s * 32 + half * 16;
      v16bf a0, a1, b0, b1;
      const __bf16* pa0 = &As[wm * 32 + row][kb];
      const __bf16* pa1 = &As[wm * 32 + 16 + row][kb];
      const __bf16* pb0 = &Bs[wn * 32 + row][kb];
      const __bf16* pb1 = &Bs[wn * 32 + 16 + row][kb];
#pragma unroll
      for (int i = 0; i < 16; ++i) {
        a0[i] = pa0[i]; a1[i] = pa1[i]; b0[i] = pb0[i]; b1[i] = pb1[i];
      }
      acc00 = __builtin_amdgcn_wmma_f32_16x16x32_bf16(false, a0, false, b0,
                                                      (short)0, acc00, false, false);
      acc01 = __builtin_amdgcn_wmma_f32_16x16x32_bf16(false, a0, false, b1,
                                                      (short)0, acc01, false, false);
      acc10 = __builtin_amdgcn_wmma_f32_16x16x32_bf16(false, a1, false, b0,
                                                      (short)0, acc10, false, false);
      acc11 = __builtin_amdgcn_wmma_f32_16x16x32_bf16(false, a1, false, b1,
                                                      (short)0, acc11, false, false);
    }
    __syncthreads();
  }

  const int gmB = m0 + wm * 32 + half * 8;
  const int cB  = n0 + wn * 32 + row;
#pragma unroll
  for (int r = 0; r < 8; ++r) {
    int gm0 = gmB + r;
    int gm1 = gmB + 16 + r;
    int c0 = cB, c1 = cB + 16;
    float b0v = (bias && c0 < N) ? bias[c0] : 0.f;
    float b1v = (bias && c1 < N) ? bias[c1] : 0.f;
    if (gm0 < M) {
      if (c0 < N) C[(long long)gm0 * ldc + c0] = apply_act(acc00[r] * alpha + b0v, act);
      if (c1 < N) C[(long long)gm0 * ldc + c1] = apply_act(acc01[r] * alpha + b1v, act);
    }
    if (gm1 < M) {
      if (c0 < N) C[(long long)gm1 * ldc + c0] = apply_act(acc10[r] * alpha + b0v, act);
      if (c1 < N) C[(long long)gm1 * ldc + c1] = apply_act(acc11[r] * alpha + b1v, act);
    }
  }
}

// ---------------------------------------------------------------------------
// Row gather: dst[b,l,:] = src[b, idx[b,l], :]
// ---------------------------------------------------------------------------
__global__ void mpsu_gather(const float* __restrict__ src, const int* __restrict__ idx,
                            float* __restrict__ dst)
{
  const int bl = blockIdx.x;            // over B*L
  const int b  = bl / L_;
  const int j  = idx[bl];
  const float* s = src + ((long long)b * L_ + j) * D_;
  float* d = dst + (long long)bl * D_;
  for (int t = threadIdx.x; t < D_; t += blockDim.x) d[t] = s[t];
}

// ---------------------------------------------------------------------------
// Fused depthwise causal conv (K=4) + SiLU, and silu(z) extraction from xz.
// ---------------------------------------------------------------------------
__global__ void mpsu_conv_silu(const float* __restrict__ xz, const float* __restrict__ cw,
                               const float* __restrict__ cb, float* __restrict__ xc,
                               float* __restrict__ zs, long long total)
{
  long long i = (long long)blockIdx.x * blockDim.x + threadIdx.x;
  if (i >= total) return;
  int c = (int)(i % DI_);
  long long bl = i / DI_;
  int l = (int)(bl % L_);
  long long b = bl / L_;
  const float* base = xz + b * (long long)L_ * (2 * DI_);
  float acc = cb[c];
#pragma unroll
  for (int k = 0; k < K_; ++k) {
    int ll = l - (K_ - 1) + k;
    if (ll >= 0) acc += base[(long long)ll * (2 * DI_) + c] * cw[c * K_ + k];
  }
  xc[i] = siluf(acc);
  zs[i] = siluf(base[(long long)l * (2 * DI_) + DI_ + c]);
}

// ---------------------------------------------------------------------------
// Selective scan: one thread per (b,d); N=8 states in registers; fuses the
// D_skip skip-connection and the silu(z) gate.
// ---------------------------------------------------------------------------
__global__ __launch_bounds__(512)
void mpsu_scan(const float* __restrict__ delta, const float* __restrict__ proj,
               const float* __restrict__ xc, const float* __restrict__ zs,
               const float* __restrict__ A_log, const float* __restrict__ D_skip,
               float* __restrict__ y)
{
  const int b = blockIdx.x;
  const int d = threadIdx.x;    // Di = 512 threads
  float A[N_], h[N_];
#pragma unroll
  for (int n = 0; n < N_; ++n) { A[n] = -__expf(A_log[d * N_ + n]); h[n] = 0.f; }
  const float dsk = D_skip[d];
  const long long base = (long long)b * L_ * DI_ + d;
  const float* pr = proj + (long long)b * L_ * (R_ + 2 * N_);
  for (int l = 0; l < L_; ++l) {
    const long long off = base + (long long)l * DI_;
    const float dt = delta[off];
    const float xt = xc[off];
    const float dx = dt * xt;
    const float* p = pr + l * (R_ + 2 * N_);
    float acc = 0.f;
#pragma unroll
    for (int n = 0; n < N_; ++n) {
      h[n] = h[n] * __expf(dt * A[n]) + dx * p[R_ + n];   // Bm
      acc += h[n] * p[R_ + N_ + n];                       // Cm
    }
    y[off] = (acc + dsk * xt) * zs[off];
  }
}

// ---------------------------------------------------------------------------
// Masked softmax over the last dim of scores (B,L,L), in place. Stages the
// score row and adjacency row through LDS with gfx1250 async-to-LDS loads
// (plain-copy fallback), all passes in LDS, single global writeback.
// ---------------------------------------------------------------------------
__global__ __launch_bounds__(256)
void mpsu_masked_softmax(float* __restrict__ s, const float* __restrict__ adj,
                         const float* __restrict__ mask)
{
  __shared__ float srow[L_];
  __shared__ float arow[L_];
  __shared__ float red[256];
  const int bi  = blockIdx.x;   // over B*L
  const int b   = bi / L_;
  const int tid = threadIdx.x;
  float* row = s + (long long)bi * L_;
  const float* arowg = adj + (long long)bi * L_;
  const float* mrow  = mask + (long long)b * L_;

#if HAVE_ASYNC_LDS
  if (tid < 128) {
    __builtin_amdgcn_global_load_async_to_lds_b128(
        AS1P(row + tid * 4), AS3P(&srow[tid * 4]), 0, 0);
  } else {
    int t = tid - 128;
    __builtin_amdgcn_global_load_async_to_lds_b128(
        AS1P(arowg + t * 4), AS3P(&arow[t * 4]), 0, 0);
  }
  __builtin_amdgcn_s_wait_asynccnt(0);
  __syncthreads();
#else
  for (int j = tid; j < L_; j += 256) { srow[j] = row[j]; arow[j] = arowg[j]; }
  __syncthreads();
#endif

  float mx = -3.4e38f;
  for (int j = tid; j < L_; j += 256) {
    bool ok = (arow[j] > 0.f) && (mrow[j] > 0.f);
    float v = ok ? srow[j] : -1e9f;
    srow[j] = v;
    mx = fmaxf(mx, v);
  }
  red[tid] = mx; __syncthreads();
  for (int st = 128; st > 0; st >>= 1) {
    if (tid < st) red[tid] = fmaxf(red[tid], red[tid + st]);
    __syncthreads();
  }
  mx = red[0]; __syncthreads();

  float sum = 0.f;
  for (int j = tid; j < L_; j += 256) {
    float e = __expf(srow[j] - mx);
    srow[j] = e;
    sum += e;
  }
  red[tid] = sum; __syncthreads();
  for (int st = 128; st > 0; st >>= 1) {
    if (tid < st) red[tid] += red[tid + st];
    __syncthreads();
  }
  const float inv = 1.f / red[0];
  __syncthreads();
  for (int j = tid; j < L_; j += 256) row[j] = srow[j] * inv;
}

// ---------------------------------------------------------------------------
// out = LayerNorm(a + b) * g + be, D=256 == blockDim.
// ---------------------------------------------------------------------------
__global__ __launch_bounds__(256)
void mpsu_add_ln(const float* __restrict__ a, const float* __restrict__ b,
                 const float* __restrict__ g, const float* __restrict__ be,
                 float* __restrict__ out)
{
  __shared__ float red[256];
  const int rowi = blockIdx.x;
  const int tid  = threadIdx.x;
  const long long off = (long long)rowi * D_ + tid;
  float v = a[off] + b[off];
  red[tid] = v; __syncthreads();
  for (int st = 128; st > 0; st >>= 1) {
    if (tid < st) red[tid] += red[tid + st];
    __syncthreads();
  }
  const float mu = red[0] / (float)D_;
  __syncthreads();
  float dv = v - mu;
  red[tid] = dv * dv; __syncthreads();
  for (int st = 128; st > 0; st >>= 1) {
    if (tid < st) red[tid] += red[tid + st];
    __syncthreads();
  }
  const float var = red[0] / (float)D_;
  out[off] = dv * rsqrtf(var + 1e-5f) * g[tid] + be[tid];
}

// ---------------------------------------------------------------------------
// Host orchestration.
// ---------------------------------------------------------------------------
extern "C" void kernel_launch(void* const* d_in, const int* in_sizes, int n_in,
                              void* d_out, int out_size, void* d_ws, size_t ws_size,
                              hipStream_t stream)
{
  const float* x      = (const float*)d_in[0];
  const float* mask   = (const float*)d_in[1];
  const float* adj    = (const float*)d_in[2];
  const int*   sidx   = (const int*)d_in[3];
  const int*   ridx   = (const int*)d_in[4];
  const float* W_in   = (const float*)d_in[5];
  const float* conv_w = (const float*)d_in[6];
  const float* conv_b = (const float*)d_in[7];
  const float* W_xp   = (const float*)d_in[8];
  const float* W_dt   = (const float*)d_in[9];
  const float* b_dt   = (const float*)d_in[10];
  const float* A_log  = (const float*)d_in[11];
  const float* D_skip = (const float*)d_in[12];
  const float* W_out  = (const float*)d_in[13];
  const float* Wq     = (const float*)d_in[14];
  const float* Wk     = (const float*)d_in[15];
  const float* Wv     = (const float*)d_in[16];
  const float* Wo     = (const float*)d_in[17];
  const float* W_view = (const float*)d_in[18];
  const float* b_view = (const float*)d_in[19];
  const float* ln1_g  = (const float*)d_in[20];
  const float* ln1_b  = (const float*)d_in[21];
  const float* W1     = (const float*)d_in[22];
  const float* b1     = (const float*)d_in[23];
  const float* W2     = (const float*)d_in[24];
  const float* b2     = (const float*)d_in[25];
  const float* ln2_g  = (const float*)d_in[26];
  const float* ln2_b  = (const float*)d_in[27];

  const long long BLD  = (long long)B_ * L_ * D_;
  const long long BLDI = (long long)B_ * L_ * DI_;
  const long long BLP  = (long long)B_ * L_ * (R_ + 2 * N_);
  const long long BLL  = (long long)B_ * L_ * L_;

  float* ws     = (float*)d_ws;
  float* concat = ws;                   // (B,L,V*D)
  float* se     = concat + BLD * V_;    // (B,L,D)
  float* xz     = se + BLD;             // (B,L,2*Di)
  float* xc     = xz + 2 * BLDI;        // (B,L,Di)
  float* zs     = xc + BLDI;            // (B,L,Di)
  float* proj   = zs + BLDI;            // (B,L,32)
  float* delta  = proj + BLP;           // (B,L,Di)
  float* ybuf   = delta + BLDI;         // (B,L,Di)
  float* mam    = ybuf + BLDI;          // (B,L,D)
  float* re     = mam + BLD;            // (B,L,D)
  float* q      = re + BLD;             // (B,L,D)
  float* k      = q + BLD;              // (B,L,D)
  float* v      = k + BLD;              // (B,L,D)
  float* sc     = v + BLD;              // (B,L,L)
  float* attn   = sc + BLL;             // (B,L,D)
  // Phase-2 aliases (per-view buffers are dead by then).
  float* yview  = mam;
  float* x1     = se;
  float* h1     = xz;
  float* y2     = xc;

  const int M = B_ * L_;   // 4096

  auto gemm = [&](const float* A, const float* Bp, float* Cp, const float* bias,
                  int m, int n, int kk, int lda, int ldb, int ldc,
                  long long sA, long long sB, long long sC, int batch,
                  int transB, float alpha, int act) {
    dim3 grid((n + BN - 1) / BN, (m + BM - 1) / BM, batch);
    const bool fast = (m % BM == 0) && (n % BN == 0) && (kk % BK == 0);
    if (fast) {
      if (transB)
        mpsu_gemm_fast<1><<<grid, dim3(256), 0, stream>>>(A, Bp, Cp, bias,
            m, n, kk, lda, ldb, ldc, sA, sB, sC, alpha, act);
      else
        mpsu_gemm_fast<0><<<grid, dim3(256), 0, stream>>>(A, Bp, Cp, bias,
            m, n, kk, lda, ldb, ldc, sA, sB, sC, alpha, act);
    } else {
      mpsu_gemm_wmma<<<grid, dim3(256), 0, stream>>>(A, Bp, Cp, bias, m, n, kk,
          lda, ldb, ldc, sA, sB, sC, transB, alpha, act);
    }
  };

  for (int i = 0; i < V_; ++i) {
    mpsu_gather<<<M, 256, 0, stream>>>(x, sidx + (long long)i * M, se);
    // xz = se @ W_in                       (4096 x 1024 x 256)  fast
    gemm(se, W_in, xz, nullptr, M, 2 * DI_, D_, D_, 2 * DI_, 2 * DI_,
         0, 0, 0, 1, 0, 1.f, 0);
    mpsu_conv_silu<<<(int)((BLDI + 255) / 256), 256, 0, stream>>>(
        xz, conv_w, conv_b, xc, zs, BLDI);
    // proj = xc @ W_xp                     (4096 x 32 x 512)    general (N=32)
    gemm(xc, W_xp, proj, nullptr, M, R_ + 2 * N_, DI_, DI_, R_ + 2 * N_,
         R_ + 2 * N_, 0, 0, 0, 1, 0, 1.f, 0);
    // delta = softplus(proj[:, :16] @ W_dt + b_dt)  (4096x512x16) general (K=16)
    gemm(proj, W_dt, delta, b_dt, M, DI_, R_, R_ + 2 * N_, DI_, DI_,
         0, 0, 0, 1, 0, 1.f, 2);
    mpsu_scan<<<B_, DI_, 0, stream>>>(delta, proj, xc, zs, A_log, D_skip, ybuf);
    // mam = ybuf @ W_out                   (4096 x 256 x 512)   fast
    gemm(ybuf, W_out, mam, nullptr, M, D_, DI_, DI_, D_, D_,
         0, 0, 0, 1, 0, 1.f, 0);
    mpsu_gather<<<M, 256, 0, stream>>>(mam, ridx + (long long)i * M, re);
    // q, k, v projections                  (4096 x 256 x 256)   fast
    gemm(re, Wq, q, nullptr, M, D_, D_, D_, D_, D_, 0, 0, 0, 1, 0, 1.f, 0);
    gemm(re, Wk, k, nullptr, M, D_, D_, D_, D_, D_, 0, 0, 0, 1, 0, 1.f, 0);
    gemm(re, Wv, v, nullptr, M, D_, D_, D_, D_, D_, 0, 0, 0, 1, 0, 1.f, 0);
    // scores = q @ k^T / 16, per batch     (8 x [512 x 512 x 256])  fast/transB
    gemm(q, k, sc, nullptr, L_, L_, D_, D_, D_, L_,
         (long long)L_ * D_, (long long)L_ * D_, (long long)L_ * L_, B_,
         1, 0.0625f, 0);
    mpsu_masked_softmax<<<M, 256, 0, stream>>>(sc, adj, mask);
    // attn = probs @ v, per batch          (8 x [512 x 256 x 512])  fast
    gemm(sc, v, attn, nullptr, L_, D_, L_, L_, D_, D_,
         BLL / B_, (long long)L_ * D_, (long long)L_ * D_, B_, 0, 1.f, 0);
    // concat[..., i*D:(i+1)*D] = attn @ Wo (4096 x 256 x 256, strided C) fast
    gemm(attn, Wo, concat + (long long)i * D_, nullptr, M, D_, D_, D_, D_,
         V_ * D_, 0, 0, 0, 1, 0, 1.f, 0);
  }

  // yview = concat @ W_view + b_view       (4096 x 256 x 2304)  fast
  gemm(concat, W_view, yview, b_view, M, D_, V_ * D_, V_ * D_, D_, D_,
       0, 0, 0, 1, 0, 1.f, 0);
  mpsu_add_ln<<<M, D_, 0, stream>>>(x, yview, ln1_g, ln1_b, x1);
  // h1 = relu(X1 @ W1 + b1)                (4096 x 1024 x 256)  fast
  gemm(x1, W1, h1, b1, M, DFF_, D_, D_, DFF_, DFF_, 0, 0, 0, 1, 0, 1.f, 1);
  // y2 = h1 @ W2 + b2                      (4096 x 256 x 1024)  fast
  gemm(h1, W2, y2, b2, M, D_, DFF_, DFF_, D_, D_, 0, 0, 0, 1, 0, 1.f, 0);
  mpsu_add_ln<<<M, D_, 0, stream>>>(x1, y2, ln2_g, ln2_b, (float*)d_out);

  (void)in_sizes; (void)n_in; (void)out_size; (void)ws_size;
}
